// GCN_65197603553732
// MI455X (gfx1250) — matrix operand order repaired and checked by reading, not compiled, not run on hardware
//
#include <hip/hip_runtime.h>
#include <hip/hip_bf16.h>

typedef __attribute__((ext_vector_type(16))) _Float16 v16h;
typedef __attribute__((ext_vector_type(8)))  float    v8f;

#define HDIM 256

// ---------------- degree / normalization ----------------
__global__ void deg_count(const int* __restrict__ dst, float* __restrict__ deg, int E) {
  int e = blockIdx.x * blockDim.x + threadIdx.x;
  if (e < E) atomicAdd(&deg[dst[e]], 1.0f);
}

__global__ void deg_to_dinv(float* __restrict__ deg, int N) {
  int i = blockIdx.x * blockDim.x + threadIdx.x;
  if (i < N) deg[i] = rsqrtf(deg[i] + 1.0f);   // +1 self loop
}

// ---------------- WMMA GEMM: D[nrows x 256] = A[nrows x K] @ W[K x 256] ----------------
// Block: 128 threads = 4 waves. Block tile: 128 rows x 64 cols.
//   - W slice (K x 64) staged to LDS once per block, pre-swizzled into WMMA B-fragment
//     order so each wave reads a v16h fragment as two ds_load_b128.
//   - A tile (16 x K) staged to LDS per row-tile in A-fragment order, shared by 4 waves.
//   - K is a template param -> kb loop fully unrolls -> straight-line v_wmma chain.
// Fragment layouts used here (wave32; A/B in contiguous k order, not ISA table order):
//   A 16x32 f16 : lane%16 = M; h = lane/16; elem e: k = kb*32 + 16h + e
//   B 32x16 f16 : lane%16 = N; h = lane/16; elem e: k = kb*32 + 16h + e
//   C/D 16x16 f32: vgpr r: M = r + 8h; N = lane%16
template <int K>
__global__ __launch_bounds__(128) void gemm_wmma(const float* __restrict__ A,
                                                 const float* __restrict__ W,
                                                 float* __restrict__ D, int nrows) {
  constexpr int KB = K / 32;
  __shared__ __align__(32) _Float16 lB[4 * KB * 32 * 16];  // [colTile][kb][lane][e]
  __shared__ __align__(32) _Float16 lA[KB * 32 * 16];      // [kb][lane][e]

  const int tid  = threadIdx.x;
  const int lane = tid & 31;
  const int wave = tid >> 5;
  const int h    = lane >> 4;
  const int r    = lane & 15;
  const int nbase = blockIdx.x * 64;
  const int col   = nbase + wave * 16 + r;

  // ---- stage B slice (K x 64) into LDS, fragment order (coalesced global reads) ----
  for (int idx = tid; idx < K * 64; idx += 128) {
    const int nl = idx & 63;          // local col 0..63
    const int k  = idx >> 6;          // 0..K-1
    const float v = W[(long)k * HDIM + nbase + nl];
    const int kb = k >> 5;
    const int hh = (k >> 4) & 1;
    const int ee = k & 15;
    const int t  = nl >> 4;
    lB[(((t * KB + kb) * 32) + hh * 16 + (nl & 15)) * 16 + ee] = (_Float16)v;
  }

  for (int mt = 0; mt < 8; ++mt) {
    const int m0 = blockIdx.y * 128 + mt * 16;   // block-uniform
    __syncthreads();                             // previous lA use done / lB fill done
    // ---- stage A tile (16 x K) into LDS, fragment order ----
    for (int idx = tid; idx < 16 * K; idx += 128) {
      const int k  = idx % K;                    // consecutive tid -> consecutive k
      const int rw = idx / K;
      int row = m0 + rw;
      if (row >= nrows) row = nrows - 1;         // clamp (tail tile)
      const float v = A[(long)row * K + k];
      const int kb  = k >> 5;
      const int r31 = k & 31;
      const int hh  = (r31 >> 4) & 1;
      const int ee  = r31 & 15;
      lA[((kb * 32) + hh * 16 + rw) * 16 + ee] = (_Float16)v;
    }
    __syncthreads();

    if (m0 < nrows) {                            // uniform scalar branch
      v8f acc = {};
#pragma unroll
      for (int kb = 0; kb < KB; ++kb) {
        const v16h a = *(const v16h*)&lA[(kb * 32 + lane) * 16];
        const v16h b = *(const v16h*)&lB[((wave * KB + kb) * 32 + lane) * 16];
        acc = __builtin_amdgcn_wmma_f32_16x16x32_f16(
            /*neg_a=*/false, a, /*neg_b=*/false, b,
            /*c_mod=*/(short)0, acc, /*reuse_a=*/false, /*reuse_b=*/false);
      }
      if (m0 + 16 <= nrows) {                    // full tile: clean store path
#pragma unroll
        for (int rr = 0; rr < 8; ++rr)
          D[(long)(m0 + rr + 8 * h) * HDIM + col] = acc[rr];
      } else {                                   // tail tile only
#pragma unroll
        for (int rr = 0; rr < 8; ++rr) {
          const int m = m0 + rr + 8 * h;
          if (m < nrows) D[(long)m * HDIM + col] = acc[rr];
        }
      }
    }
  }
}

// ---------------- edge scatter-add (the bandwidth-bound part) ----------------
// 4 edges per 256-thread block; each thread: one float4 gather + 4 f32 atomics
__global__ void scatter_edges(const int* __restrict__ src, const int* __restrict__ dst,
                              const float* __restrict__ dinv, const float* __restrict__ hfeat,
                              float* __restrict__ agg, int E) {
  const int e = blockIdx.x * 4 + (threadIdx.x >> 6);
  if (e >= E) return;
  const int s = src[e];
  const int d = dst[e];
  const float c = dinv[s] * dinv[d];
  const int f = (threadIdx.x & 63) * 4;
  const float4 hv = *(const float4*)(hfeat + (long)s * HDIM + f);
  float* ap = agg + (long)d * HDIM + f;
  atomicAdd(ap + 0, hv.x * c);
  atomicAdd(ap + 1, hv.y * c);
  atomicAdd(ap + 2, hv.z * c);
  atomicAdd(ap + 3, hv.w * c);
}

// ---------------- self-loop + bias + ReLU, in place on agg (float4) ----------------
__global__ void post_selfloop_bias_relu(float* __restrict__ agg, const float* __restrict__ hfeat,
                                        const float* __restrict__ dinv,
                                        const float* __restrict__ bias, long total4) {
  const long q = (long)blockIdx.x * blockDim.x + threadIdx.x;   // index in float4 units
  if (q >= total4) return;
  const long idx = q * 4;
  const int i = (int)(idx >> 8);            // node
  const int f = (int)(idx & (HDIM - 1));
  const float di = dinv[i];
  const float d2 = di * di;
  const float4 av = *(const float4*)(agg + idx);
  const float4 hv = *(const float4*)(hfeat + idx);
  const float4 bv = *(const float4*)(bias + f);
  float4 o;
  o.x = fmaxf(av.x + hv.x * d2 + bv.x, 0.0f);
  o.y = fmaxf(av.y + hv.y * d2 + bv.y, 0.0f);
  o.z = fmaxf(av.z + hv.z * d2 + bv.z, 0.0f);
  o.w = fmaxf(av.w + hv.w * d2 + bv.w, 0.0f);
  *(float4*)(agg + idx) = o;
}

// ---------------- per-feature sum over nodes ----------------
__global__ void reduce_nodes(const float* __restrict__ hfeat, float* __restrict__ gsum, int N) {
  const int f = threadIdx.x;                   // 0..255
  const long i0 = (long)blockIdx.x * 64;
  float s = 0.0f;
  for (int i = 0; i < 64; ++i) {
    const long node = i0 + i;
    if (node < N) s += hfeat[node * HDIM + f];
  }
  atomicAdd(&gsum[f], s);
}

// ---------------- final 256 -> 2 linear on the node-mean ----------------
__global__ void final_linear(const float* __restrict__ gsum, const float* __restrict__ Wl,
                             const float* __restrict__ bl, float* __restrict__ out, int N) {
  __shared__ float s0[HDIM];
  __shared__ float s1[HDIM];
  const int f = threadIdx.x;
  const float v = gsum[f] * (1.0f / (float)N);
  s0[f] = v * Wl[f * 2 + 0];
  s1[f] = v * Wl[f * 2 + 1];
  __syncthreads();
  for (int stride = HDIM / 2; stride > 0; stride >>= 1) {
    if (f < stride) { s0[f] += s0[f + stride]; s1[f] += s1[f + stride]; }
    __syncthreads();
  }
  if (f == 0) {
    out[0] = s0[0] + bl[0];
    out[1] = s1[0] + bl[1];
  }
}

extern "C" void kernel_launch(void* const* d_in, const int* in_sizes, int n_in,
                              void* d_out, int out_size, void* d_ws, size_t ws_size,
                              hipStream_t stream) {
  const float* x  = (const float*)d_in[0];
  const int*   ei = (const int*)d_in[1];   // [2, E] row-major
  // d_in[2] = batch (unused; gives N)
  const float* W1 = (const float*)d_in[3];
  const float* b1 = (const float*)d_in[4];
  const float* W2 = (const float*)d_in[5];
  const float* b2 = (const float*)d_in[6];
  const float* Wl = (const float*)d_in[7];
  const float* bl = (const float*)d_in[8];
  float* out = (float*)d_out;

  const int N    = in_sizes[2];
  const int E    = in_sizes[1] / 2;
  const int F_IN = in_sizes[0] / N;        // 128
  const int* src = ei;
  const int* dst = ei + E;

  // workspace layout (256B aligned)
  auto al = [](size_t v) { return (v + 255) & ~(size_t)255; };
  char* ws = (char*)d_ws;
  size_t off = 0;
  float* dinv = (float*)(ws + off); off += al((size_t)N * 4);
  float* gsum = (float*)(ws + off); off += al((size_t)HDIM * 4);
  float* bufA = (float*)(ws + off); off += al((size_t)N * HDIM * 4);
  float* bufB = (float*)(ws + off); off += al((size_t)N * HDIM * 4);
  (void)ws_size;

  const long total  = (long)N * HDIM;
  const long total4 = total / 4;
  const dim3 gemmGrid(HDIM / 64, (N + 127) / 128);   // 64-col groups x 128-row groups
  const int  postBlocks = (int)((total4 + 255) / 256);
  const int  scatBlocks = (E + 3) / 4;

  // degrees -> dinv (in place)
  hipMemsetAsync(dinv, 0, (size_t)N * 4, stream);
  deg_count<<<(E + 255) / 256, 256, 0, stream>>>(dst, dinv, E);
  deg_to_dinv<<<(N + 255) / 256, 256, 0, stream>>>(dinv, N);

  // ---- layer 1: bufA = x @ W1 ; bufB = aggregate ----
  if (F_IN == 128) gemm_wmma<128><<<gemmGrid, 128, 0, stream>>>(x, W1, bufA, N);
  else             gemm_wmma<256><<<gemmGrid, 128, 0, stream>>>(x, W1, bufA, N);
  hipMemsetAsync(bufB, 0, (size_t)total * 4, stream);
  scatter_edges<<<scatBlocks, 256, 0, stream>>>(src, dst, dinv, bufA, bufB, E);
  post_selfloop_bias_relu<<<postBlocks, 256, 0, stream>>>(bufB, bufA, dinv, b1, total4);

  // ---- layer 2: bufA = bufB @ W2 ; reuse bufB as aggregate ----
  gemm_wmma<256><<<gemmGrid, 128, 0, stream>>>(bufB, W2, bufA, N);
  hipMemsetAsync(bufB, 0, (size_t)total * 4, stream);
  scatter_edges<<<scatBlocks, 256, 0, stream>>>(src, dst, dinv, bufA, bufB, E);
  post_selfloop_bias_relu<<<postBlocks, 256, 0, stream>>>(bufB, bufA, dinv, b2, total4);

  // ---- global mean pool + final linear ----
  hipMemsetAsync(gsum, 0, (size_t)HDIM * 4, stream);
  reduce_nodes<<<(N + 63) / 64, HDIM, 0, stream>>>(bufB, gsum, N);
  final_linear<<<1, HDIM, 0, stream>>>(gsum, Wl, bl, out, N);
}
